// MOLERouterV3_45586782880337
// MI455X (gfx1250) — matrix operand order repaired and checked by reading, not compiled, not run on hardware
//
#include <hip/hip_runtime.h>
#include <math.h>

#define B_ROWS 131072
#define IN_DIM 2048
#define H_DIM  128
#define E_DIM  64
#define TOPK   8
#define TILE_M 128
#define NCHUNK (IN_DIM / 32)
#define EPSF   1e-8f

typedef __attribute__((ext_vector_type(16))) __bf16          v16bf;
typedef __attribute__((ext_vector_type(8)))  float           v8f;
typedef __attribute__((ext_vector_type(8)))  unsigned int    v8ui;
typedef __attribute__((ext_vector_type(8)))  unsigned short  v8us;
typedef __attribute__((ext_vector_type(16))) unsigned short  v16us;

#if __has_builtin(__builtin_amdgcn_perm)
#define HAVE_PERM 1
#else
#define HAVE_PERM 0
#endif

// float -> bf16, round-to-nearest (+0x8000), scalar. 2 VALU ops.
static __device__ __forceinline__ unsigned short f2bf(float f) {
    return (unsigned short)((__float_as_uint(f) + 0x8000u) >> 16);
}

// two floats -> one dword holding {bf16(fb), bf16(fa)}: 2 adds + 1 v_perm_b32.
static __device__ __forceinline__ unsigned int packbf2(float fa, float fb) {
    unsigned int ua = __float_as_uint(fa) + 0x8000u;
    unsigned int ub = __float_as_uint(fb) + 0x8000u;
#if HAVE_PERM
    // dst bytes: [0]=ua.b2 [1]=ua.b3 [2]=ub.b2 [3]=ub.b3  (S1=ua -> sel 2,3; S0=ub -> sel 6,7)
    return __builtin_amdgcn_perm(ub, ua, 0x07060302u);
#else
    return (ua >> 16) | (ub & 0xFFFF0000u);
#endif
}

// Build a 16-element bf16 fragment from two 8-element runs in LDS.
static __device__ __forceinline__ v16bf pack2x8(const unsigned short* p0,
                                                const unsigned short* p1) {
    v8us lo = *(const v8us*)p0;
    v8us hi = *(const v8us*)p1;
    v16us w = __builtin_shufflevector(lo, hi, 0,1,2,3,4,5,6,7,8,9,10,11,12,13,14,15);
    return __builtin_bit_cast(v16bf, w);
}

// Convert 16 f32 (4x float4) to a bf16 A-fragment, built as 8 packed dwords.
static __device__ __forceinline__ v16bf cvtA16(float4 f0, float4 f1,
                                               float4 f2, float4 f3) {
    v8ui r;
    r[0] = packbf2(f0.x, f0.y);
    r[1] = packbf2(f0.z, f0.w);
    r[2] = packbf2(f1.x, f1.y);
    r[3] = packbf2(f1.z, f1.w);
    r[4] = packbf2(f2.x, f2.y);
    r[5] = packbf2(f2.z, f2.w);
    r[6] = packbf2(f3.x, f3.y);
    r[7] = packbf2(f3.z, f3.w);
    return __builtin_bit_cast(v16bf, r);
}

// A-fragment (16x32 bf16 layout) straight from a global f32 row.
static __device__ __forceinline__ v16bf loadA(const float* __restrict__ xrow,
                                              int kbase) {
    float4 f0 = *(const float4*)(xrow + kbase);
    float4 f1 = *(const float4*)(xrow + kbase + 4);
    float4 f2 = *(const float4*)(xrow + kbase + 16);
    float4 f3 = *(const float4*)(xrow + kbase + 20);
    return cvtA16(f0, f1, f2, f3);
}

// Stage one 32-K chunk of W1 into LDS, transposed [n][k], bf16.
// Coalesced float4 row reads; transpose happens on the LDS-store side.
static __device__ __forceinline__ void stage_w1_chunk(
    const float* __restrict__ W1, int c, int tid, unsigned short* dst) {
    const int k  = tid >> 3;          // 0..31 : k within chunk
    const int n0 = (tid & 7) * 16;    // 0..112: 16 consecutive n per thread
    const float* src = W1 + (size_t)(c * 32 + k) * H_DIM + n0;
    float4 a0 = *(const float4*)(src);
    float4 a1 = *(const float4*)(src + 4);
    float4 a2 = *(const float4*)(src + 8);
    float4 a3 = *(const float4*)(src + 12);
    unsigned short u[16];
    u[0]  = f2bf(a0.x); u[1]  = f2bf(a0.y); u[2]  = f2bf(a0.z); u[3]  = f2bf(a0.w);
    u[4]  = f2bf(a1.x); u[5]  = f2bf(a1.y); u[6]  = f2bf(a1.z); u[7]  = f2bf(a1.w);
    u[8]  = f2bf(a2.x); u[9]  = f2bf(a2.y); u[10] = f2bf(a2.z); u[11] = f2bf(a2.w);
    u[12] = f2bf(a3.x); u[13] = f2bf(a3.y); u[14] = f2bf(a3.z); u[15] = f2bf(a3.w);
    #pragma unroll
    for (int j = 0; j < 16; ++j)
        dst[(n0 + j) * 32 + k] = u[j];
}

__global__ void moe_router_zero_ws(unsigned int* ws) {
    if (threadIdx.x < 65) ws[threadIdx.x] = 0u;   // 64 counts + 1 float monitor accum
}

__global__ __launch_bounds__(256) void moe_router_main(
    const float* __restrict__ X,  const float* __restrict__ W1,
    const float* __restrict__ b1, const float* __restrict__ W2,
    const float* __restrict__ b2, const float* __restrict__ ebias,
    float* __restrict__ out, unsigned int* __restrict__ counts,
    float* __restrict__ monitor_accum)
{
    __shared__ unsigned short w1t[2][H_DIM * 32];   // double-buffered W1 chunk (2x8 KB)
    __shared__ unsigned short w2t[E_DIM * H_DIM];   // W2 transposed bf16 (16 KB)
    __shared__ unsigned short hlds[TILE_M * H_DIM]; // h bf16; later overlaid by scores f32
    __shared__ float b1s[H_DIM];
    __shared__ float b2s[E_DIM];
    __shared__ float ebs[E_DIM];
    __shared__ int   hist[E_DIM];
    __shared__ float monred;

    const int tid   = threadIdx.x;
    const int wave  = tid >> 5;
    const int lane  = tid & 31;
    const int lhalf = (lane < 16) ? 0 : 1;
    const int l16   = lane & 15;
    const int r0    = blockIdx.x * TILE_M;

    // ---- stage W2 transposed bf16, biases, zero reductions ----
    {
        int n  = tid >> 2;           // 0..63
        int kb = (tid & 3) * 32;
        #pragma unroll 8
        for (int j = 0; j < 32; ++j)
            w2t[n * H_DIM + kb + j] = f2bf(W2[(size_t)(kb + j) * E_DIM + n]);
    }
    if (tid < H_DIM) b1s[tid] = b1[tid];
    if (tid < E_DIM) { b2s[tid] = b2[tid]; ebs[tid] = ebias[tid]; hist[tid] = 0; }
    if (tid == 0) monred = 0.f;

    // ---- GEMM1: [128 x 2048] @ [2048 x 128], bf16 WMMA, f32 accum ----
    v8f acc1[8];
    #pragma unroll
    for (int nb = 0; nb < 8; ++nb) acc1[nb] = (v8f){0.f,0.f,0.f,0.f,0.f,0.f,0.f,0.f};

    const float* xrow = X + (size_t)(r0 + wave * 16 + l16) * IN_DIM; // A row: M = lane&15
    const int k0a = lhalf * 8;    // A-layout K runs: k0..k0+7, k0+16..k0+23
    const int kbB = lhalf * 8;    // B-layout K runs: k0..k0+7, k0+16..k0+23 (same as A)

    // prologue: stage chunk 0, load A fragment for chunk 0
    stage_w1_chunk(W1, 0, tid, &w1t[0][0]);
    __syncthreads();
    v16bf a = loadA(xrow, k0a);

    for (int c = 0; c < NCHUNK; ++c) {
        const int cur = c & 1;
        const bool have_next = (c + 1 < NCHUNK);

        // kick off next chunk's global traffic early (overlaps with WMMA below)
        if (have_next)
            stage_w1_chunk(W1, c + 1, tid, &w1t[cur ^ 1][0]);
        v16bf anext = a;
        if (have_next)
            anext = loadA(xrow, (c + 1) * 32 + k0a);
        if (c + 2 < NCHUNK)
            __builtin_prefetch(xrow + (c + 2) * 32 + k0a, 0, 0);

        // preload all 8 B fragments, then run the 8 WMMAs back-to-back
        v16bf bfr[8];
        #pragma unroll
        for (int nb = 0; nb < 8; ++nb) {
            const unsigned short* bp = &w1t[cur][(nb * 16 + l16) * 32 + kbB];
            bfr[nb] = pack2x8(bp, bp + 16);
        }
        #pragma unroll
        for (int nb = 0; nb < 8; ++nb)
            acc1[nb] = __builtin_amdgcn_wmma_f32_16x16x32_bf16(
                false, a, false, bfr[nb], (short)0, acc1[nb], false, false);

        a = anext;
        __syncthreads();
    }

    // ---- bias + SiLU, write h tile to LDS (own-wave rows, bf16) ----
    #pragma unroll
    for (int nb = 0; nb < 8; ++nb) {
        #pragma unroll
        for (int r = 0; r < 8; ++r) {
            int m = wave * 16 + r + lhalf * 8;      // C layout: VGPR r -> M=r / r+8
            int n = nb * 16 + l16;
            float x = acc1[nb][r] + b1s[n];
            float h = x / (1.f + __expf(-x));       // silu
            hlds[m * H_DIM + n] = f2bf(h);
        }
    }

    // ---- GEMM2: per-wave [16 x 128] @ [128 x 64] ----
    v8f acc2[4];
    #pragma unroll
    for (int nb = 0; nb < 4; ++nb) acc2[nb] = (v8f){0.f,0.f,0.f,0.f,0.f,0.f,0.f,0.f};

    #pragma unroll
    for (int c = 0; c < H_DIM / 32; ++c) {
        const int kbase = c * 32;
        const unsigned short* ap = &hlds[(wave * 16 + l16) * H_DIM + kbase + k0a];
        v16bf a2 = pack2x8(ap, ap + 16);
        v16bf bfr[4];
        #pragma unroll
        for (int nb = 0; nb < 4; ++nb) {
            const unsigned short* bp = &w2t[(nb * 16 + l16) * H_DIM + kbase + kbB];
            bfr[nb] = pack2x8(bp, bp + 16);
        }
        #pragma unroll
        for (int nb = 0; nb < 4; ++nb)
            acc2[nb] = __builtin_amdgcn_wmma_f32_16x16x32_bf16(
                false, a2, false, bfr[nb], (short)0, acc2[nb], false, false);
    }

    // ---- sigmoid -> scores, overlay own-wave rows of hlds (256B/row each way) ----
    float* scores = (float*)hlds;
    #pragma unroll
    for (int nb = 0; nb < 4; ++nb) {
        #pragma unroll
        for (int r = 0; r < 8; ++r) {
            int m = wave * 16 + r + lhalf * 8;
            int n = nb * 16 + l16;
            float x = acc2[nb][r] + b2s[n];
            scores[m * E_DIM + n] = 1.f / (1.f + __expf(-x));
        }
    }
    __syncthreads();

    // ---- per-row top-8, normalize, scatter dense coeffs, stats ----
    if (tid < TILE_M) {
        float s[E_DIM];   // original scores (for gather/normalize)
        float sb[E_DIM];  // biased scores (for selection), registers only
        #pragma unroll
        for (int e = 0; e < E_DIM; ++e) {
            s[e]  = scores[tid * E_DIM + e];
            sb[e] = s[e] + ebs[e];
        }

        int   idxs[TOPK];
        float vals[TOPK];
        float sum = 0.f;
        #pragma unroll
        for (int it = 0; it < TOPK; ++it) {
            float best = -3.4e38f; int bi = 0;
            #pragma unroll
            for (int e = 0; e < E_DIM; ++e) {
                if (sb[e] > best) { best = sb[e]; bi = e; } // strict > : lowest idx on ties
            }
            sb[bi] = -3.4e38f;                              // remove from candidates
            idxs[it] = bi;
            vals[it] = s[bi];                               // gather unbiased score
            sum += s[bi];
            atomicAdd(&hist[bi], 1);
        }
        float inv = 1.f / (sum + EPSF);

        size_t gbase = (size_t)(r0 + tid) * E_DIM;
        float4 z = {0.f, 0.f, 0.f, 0.f};
        #pragma unroll
        for (int q = 0; q < E_DIM / 4; ++q) *(float4*)(out + gbase + q * 4) = z;
        float pm = 0.f;
        #pragma unroll
        for (int it = 0; it < TOPK; ++it) {
            float p = vals[it] * inv;
            out[gbase + idxs[it]] = p;
            pm = fmaxf(pm, p);
        }
        atomicAdd(&monred, pm);
    }
    __syncthreads();

    if (tid < E_DIM && hist[tid] > 0) atomicAdd(&counts[tid], (unsigned)hist[tid]);
    if (tid == 0) atomicAdd(monitor_accum, monred);
}

__global__ void moe_router_finalize(const unsigned int* __restrict__ counts,
                                    const float* __restrict__ monitor_accum,
                                    float* __restrict__ out_scalars)
{
    if (threadIdx.x == 0) {
        float sum = 0.f;
        for (int e = 0; e < E_DIM; ++e) sum += (float)counts[e];
        float mean = sum / (float)E_DIM;
        float var = 0.f;
        for (int e = 0; e < E_DIM; ++e) {
            float d = (float)counts[e] - mean;
            var += d * d;
        }
        var /= (float)(E_DIM - 1);                          // ddof=1
        out_scalars[0] = monitor_accum[0] / (float)B_ROWS;  // monitor_val
        out_scalars[1] = sqrtf(var) / (mean + EPSF);        // expert_load_cv
    }
}

extern "C" void kernel_launch(void* const* d_in, const int* in_sizes, int n_in,
                              void* d_out, int out_size, void* d_ws, size_t ws_size,
                              hipStream_t stream) {
    const float* X  = (const float*)d_in[0];
    const float* W1 = (const float*)d_in[1];
    const float* b1 = (const float*)d_in[2];
    const float* W2 = (const float*)d_in[3];
    const float* b2 = (const float*)d_in[4];
    const float* eb = (const float*)d_in[5];
    float* out = (float*)d_out;

    unsigned int* counts = (unsigned int*)d_ws;
    float* mon = ((float*)d_ws) + E_DIM;

    moe_router_zero_ws<<<1, 128, 0, stream>>>(counts);
    moe_router_main<<<B_ROWS / TILE_M, 256, 0, stream>>>(X, W1, b1, W2, b2, eb,
                                                         out, counts, mon);
    moe_router_finalize<<<1, 64, 0, stream>>>(counts, mon,
                                              out + (size_t)B_ROWS * E_DIM);
}
